// MLA_75471165325326
// MI455X (gfx1250) — compile-verified, matrix operand order checked by reading, not run on hardware
//
#include <hip/hip_runtime.h>
#include <hip/hip_bf16.h>
#include <stdint.h>

// ---------------------------------------------------------------------------
// MLA forward for MI455X (gfx1250): bf16 WMMA everywhere, f32 accumulate.
// cvt -> QKV GEMMs (scale folded into Q) -> latent down/up GEMMs
//     -> flash attention (64-key blocks) -> out GEMM
// GEMMs are double-buffered in LDS (1 barrier / K-step, global prefetch into
// registers overlapping the wmma work).
// ---------------------------------------------------------------------------

typedef __attribute__((ext_vector_type(16))) __bf16 v16bf;
typedef __attribute__((ext_vector_type(8)))  float  v8f;

union FragBF {
    v16bf v;
    uint4 q[2];
};

union U8x16 {
    uint4 q;
    unsigned short s[8];
};

__device__ __forceinline__ unsigned short f2bf(float f) {
    union { float f; uint32_t u; } v; v.f = f;
    uint32_t r = v.u + 0x7FFFu + ((v.u >> 16) & 1u);   // round-to-nearest-even
    return (unsigned short)(r >> 16);
}

// ---------------------------------------------------------------------------
// fp32 -> bf16 conversion (grid-stride)
// ---------------------------------------------------------------------------
__global__ void cvt_f32_to_bf16(const float* __restrict__ in,
                                unsigned short* __restrict__ out, int n) {
    int i = blockIdx.x * blockDim.x + threadIdx.x;
    int stride = gridDim.x * blockDim.x;
    for (; i < n; i += stride) out[i] = f2bf(in[i]);
}

// ---------------------------------------------------------------------------
// Double-buffered bf16 GEMM:  C[M,N] = (A[M,K] * B[K,N] + bias[N]) * outScale
//   A, B row-major bf16 (ushort); f32 accumulate via v_wmma_f32_16x16x32_bf16.
//   B tiles stored transposed in LDS -> both operand fragments are b128 loads
//   matching the 16-bit A/B lane layout (lanes 0-15: K 0-7 & 16-23, lanes
//   16-31: K 8-15 & 24-31).  All dims are exact multiples of the tiles.
// ---------------------------------------------------------------------------
template <int BM, int BN, int WM, int WN, bool OUT_BF16>
__global__ void __launch_bounds__((BM / WM) * (BN / WN) * 32)
gemm_bf16_wmma(const unsigned short* __restrict__ A,
               const unsigned short* __restrict__ Bm,
               const float* __restrict__ bias,
               void* __restrict__ Cout,
               float outScale,
               int M, int N, int K) {
    constexpr int WAVES_N = BN / WN;
    constexpr int NW = (BM / WM) * (BN / WN);
    constexpr int NT = NW * 32;
    constexpr int BK = 32;
    constexpr int LD = BK + 8;            // 40: pad keeps 16B align, avoids conflicts
    constexpr int MI = WM / 16;
    constexpr int NI = WN / 16;
    constexpr int AREP = (BM * BK / 8) / NT;     // b128 chunks per thread (A tile)
    constexpr int BREP = (BK * BN) / (NT * 8);   // b128 chunks per thread (B tile)
    static_assert((BM * BK / 8) % NT == 0 && (BK * BN) % (NT * 8) == 0, "tiling");

    __shared__ unsigned short As[2][BM * LD];
    __shared__ unsigned short Bs[2][BN * LD];    // transposed: Bs[buf][n][k]

    const int tid  = threadIdx.x;
    const int lane = tid & 31;
    const int wave = tid >> 5;
    const int wm   = wave / WAVES_N;
    const int wn   = wave % WAVES_N;
    const int rowBase = blockIdx.y * BM;
    const int colBase = blockIdx.x * BN;

    const int l15     = lane & 15;
    const int halfsel = (lane < 16) ? 0 : 8;   // 16-bit A/B K-group select
    const int rowOff  = (lane < 16) ? 0 : 8;   // f32 D-layout row select

    v8f acc[MI][NI];
#pragma unroll
    for (int i = 0; i < MI; ++i)
#pragma unroll
        for (int j = 0; j < NI; ++j) acc[i][j] = v8f{};

    // ---- prologue: tile kb=0 straight into buffer 0 -----------------------
#pragma unroll
    for (int rep = 0; rep < AREP; ++rep) {
        int c  = tid + rep * NT;
        int r  = c >> 2;
        int cc = (c & 3) * 8;
        *(uint4*)&As[0][r * LD + cc] =
            *(const uint4*)&A[(size_t)(rowBase + r) * K + cc];
    }
#pragma unroll
    for (int rep = 0; rep < BREP; ++rep) {
        int e = (tid + rep * NT) * 8;
        int k = e / BN;
        int n = e % BN;
        U8x16 u;
        u.q = *(const uint4*)&Bm[(size_t)k * N + colBase + n];
#pragma unroll
        for (int j = 0; j < 8; ++j) Bs[0][(n + j) * LD + k] = u.s[j];
    }
    __syncthreads();

    // ---- main loop: prefetch next tile into regs while wmma runs ----------
    int buf = 0;
    for (int kb = 0; kb < K; kb += BK, buf ^= 1) {
        const int nxt = kb + BK;
        uint4 aPre[AREP];
        U8x16 bPre[BREP];
        if (nxt < K) {
#pragma unroll
            for (int rep = 0; rep < AREP; ++rep) {
                int c  = tid + rep * NT;
                int r  = c >> 2;
                int cc = (c & 3) * 8;
                aPre[rep] = *(const uint4*)&A[(size_t)(rowBase + r) * K + nxt + cc];
            }
#pragma unroll
            for (int rep = 0; rep < BREP; ++rep) {
                int e = (tid + rep * NT) * 8;
                int k = e / BN;
                int n = e % BN;
                bPre[rep].q = *(const uint4*)&Bm[(size_t)(nxt + k) * N + colBase + n];
            }
        }

        FragBF af[MI];
#pragma unroll
        for (int i = 0; i < MI; ++i) {
            int r = wm * WM + i * 16 + l15;
            af[i].q[0] = *(const uint4*)&As[buf][r * LD + halfsel];
            af[i].q[1] = *(const uint4*)&As[buf][r * LD + halfsel + 16];
        }
        FragBF bf_[NI];
#pragma unroll
        for (int j = 0; j < NI; ++j) {
            int c = wn * WN + j * 16 + l15;
            bf_[j].q[0] = *(const uint4*)&Bs[buf][c * LD + halfsel];
            bf_[j].q[1] = *(const uint4*)&Bs[buf][c * LD + halfsel + 16];
        }
#pragma unroll
        for (int i = 0; i < MI; ++i)
#pragma unroll
            for (int j = 0; j < NI; ++j)
                acc[i][j] = __builtin_amdgcn_wmma_f32_16x16x32_bf16(
                    false, af[i].v, false, bf_[j].v, (short)0, acc[i][j],
                    false, false);

        if (nxt < K) {
#pragma unroll
            for (int rep = 0; rep < AREP; ++rep) {
                int c  = tid + rep * NT;
                int r  = c >> 2;
                int cc = (c & 3) * 8;
                *(uint4*)&As[buf ^ 1][r * LD + cc] = aPre[rep];
            }
#pragma unroll
            for (int rep = 0; rep < BREP; ++rep) {
                int e = (tid + rep * NT) * 8;
                int k = e / BN;
                int n = e % BN;
#pragma unroll
                for (int j = 0; j < 8; ++j)
                    Bs[buf ^ 1][(n + j) * LD + k] = bPre[rep].s[j];
            }
            __syncthreads();
        }
    }

    // ---- epilogue: D-layout store + bias + scale --------------------------
#pragma unroll
    for (int i = 0; i < MI; ++i)
#pragma unroll
        for (int j = 0; j < NI; ++j)
#pragma unroll
            for (int r = 0; r < 8; ++r) {
                int gr = rowBase + wm * WM + i * 16 + rowOff + r;
                int gc = colBase + wn * WN + j * 16 + l15;
                float val = (acc[i][j][r] + bias[gc]) * outScale;
                if (OUT_BF16)
                    ((unsigned short*)Cout)[(size_t)gr * N + gc] = f2bf(val);
                else
                    ((float*)Cout)[(size_t)gr * N + gc] = val;
            }
}

// ---------------------------------------------------------------------------
// Flash attention, full (non-causal).  One workgroup = 4 waves = 64 q-rows of
// one (b,h).  64 keys per iteration: 8 wmma for S = Q*K^T, online softmax
// (stats reduced over 16-lane halves), 8 wmma for O += P*V.  The 1/sqrt(D)
// scale is pre-folded into Q by the Q-projection GEMM.
// Q/K/V/O are bf16 [B,T,C]; head h occupies columns [h*64, h*64+64).
// ---------------------------------------------------------------------------
__global__ void __launch_bounds__(128)
mla_attention(const unsigned short* __restrict__ Qg,
              const unsigned short* __restrict__ Kg,
              const unsigned short* __restrict__ Vg,
              unsigned short* __restrict__ Og) {
    constexpr int T = 2048, C = 1024, D = 64;
    constexpr int QB = 64;             // q rows per workgroup
    constexpr int KT = 64;             // keys per iteration
    constexpr int KL = 72;             // padded row stride (16B aligned: 144B)

    __shared__ unsigned short Ks[KT * KL];      // [key][d]
    __shared__ unsigned short Vs[D * KL];       // [d][key]   (transposed)
    __shared__ unsigned short Ps[4 * 16 * KL];  // per-wave P tile [row][key]

    const int tid  = threadIdx.x;
    const int lane = tid & 31;
    const int wave = tid >> 5;
    const int l15     = lane & 15;
    const int halfsel = (lane < 16) ? 0 : 8;
    const int rowOff  = (lane < 16) ? 0 : 8;

    const int qBlocks = T / QB;
    const int qb = blockIdx.x % qBlocks;
    const int bh = blockIdx.x / qBlocks;
    const int b  = bh >> 4;
    const int h  = bh & 15;
    const int qRow0 = qb * QB + wave * 16;

    // Q fragments (16 rows x 64), resident for the whole key loop
    FragBF qf[2];
    {
        size_t base = ((size_t)b * T + qRow0 + l15) * C + (size_t)h * D;
#pragma unroll
        for (int s = 0; s < 2; ++s) {
            qf[s].q[0] = *(const uint4*)&Qg[base + s * 32 + halfsel];
            qf[s].q[1] = *(const uint4*)&Qg[base + s * 32 + halfsel + 16];
        }
    }

    v8f o[4];
#pragma unroll
    for (int j = 0; j < 4; ++j) o[j] = v8f{};
    float mrow[8], lrow[8];
#pragma unroll
    for (int r = 0; r < 8; ++r) { mrow[r] = -3.0e38f; lrow[r] = 0.0f; }

    for (int kb = 0; kb < T; kb += KT) {
        __syncthreads();
        // K tile: 64 keys x 64, row-major b128 copies (acts as B^T for S)
        for (int c = tid; c < KT * D / 8; c += 128) {
            int key = c >> 3;
            int cc  = (c & 7) * 8;
            *(uint4*)&Ks[key * KL + cc] =
                *(const uint4*)&Kg[((size_t)b * T + kb + key) * C + (size_t)h * D + cc];
        }
        // V tile: b128 global reads, transposed scalar LDS stores -> Vs[d][key]
        for (int c = tid; c < KT * D / 8; c += 128) {
            int key = c >> 3;
            int d0  = (c & 7) * 8;
            U8x16 u;
            u.q = *(const uint4*)&Vg[((size_t)b * T + kb + key) * C + (size_t)h * D + d0];
#pragma unroll
            for (int j = 0; j < 8; ++j) Vs[(d0 + j) * KL + key] = u.s[j];
        }
        __syncthreads();

        // S = Q * K^T : four 16-key column tiles, two K-dim steps each
        v8f s[4];
#pragma unroll
        for (int n = 0; n < 4; ++n) {
            s[n] = v8f{};
            int row = (n * 16 + l15) * KL;
            FragBF kf;
#pragma unroll
            for (int ks = 0; ks < 2; ++ks) {
                kf.q[0] = *(const uint4*)&Ks[row + ks * 32 + halfsel];
                kf.q[1] = *(const uint4*)&Ks[row + ks * 32 + halfsel + 16];
                s[n] = __builtin_amdgcn_wmma_f32_16x16x32_bf16(
                    false, qf[ks].v, false, kf.v, (short)0, s[n], false, false);
            }
        }

        // Online softmax over the 64 new keys
        unsigned short* Pw = &Ps[wave * 16 * KL];
        float alpha[8];
#pragma unroll
        for (int r = 0; r < 8; ++r) {
            float t0 = s[0][r], t1 = s[1][r], t2 = s[2][r], t3 = s[3][r];
            float v = fmaxf(fmaxf(t0, t1), fmaxf(t2, t3));
            v = fmaxf(v, __shfl_xor(v, 1, 32));
            v = fmaxf(v, __shfl_xor(v, 2, 32));
            v = fmaxf(v, __shfl_xor(v, 4, 32));
            v = fmaxf(v, __shfl_xor(v, 8, 32));
            float mn = fmaxf(mrow[r], v);
            alpha[r] = __expf(mrow[r] - mn);
            float p0 = __expf(t0 - mn);
            float p1 = __expf(t1 - mn);
            float p2 = __expf(t2 - mn);
            float p3 = __expf(t3 - mn);
            int pr = (rowOff + r) * KL;
            Pw[pr + l15]      = f2bf(p0);
            Pw[pr + 16 + l15] = f2bf(p1);
            Pw[pr + 32 + l15] = f2bf(p2);
            Pw[pr + 48 + l15] = f2bf(p3);
            float rs = (p0 + p1) + (p2 + p3);
            rs += __shfl_xor(rs, 1, 32);
            rs += __shfl_xor(rs, 2, 32);
            rs += __shfl_xor(rs, 4, 32);
            rs += __shfl_xor(rs, 8, 32);
            lrow[r] = lrow[r] * alpha[r] + rs;
            mrow[r] = mn;
        }
#pragma unroll
        for (int j = 0; j < 4; ++j)
#pragma unroll
            for (int r = 0; r < 8; ++r) o[j][r] *= alpha[r];

        // P (16x64 bf16, wave-private) back as A fragments
        asm volatile("s_wait_dscnt 0" ::: "memory");  // order in-wave DS store->load
        FragBF pf[2];
#pragma unroll
        for (int ks = 0; ks < 2; ++ks) {
            pf[ks].q[0] = *(const uint4*)&Pw[l15 * KL + ks * 32 + halfsel];
            pf[ks].q[1] = *(const uint4*)&Pw[l15 * KL + ks * 32 + halfsel + 16];
        }

        // O += P * V
#pragma unroll
        for (int j = 0; j < 4; ++j) {
            int d = (j * 16 + l15) * KL;
            FragBF vf;
#pragma unroll
            for (int ks = 0; ks < 2; ++ks) {
                vf.q[0] = *(const uint4*)&Vs[d + ks * 32 + halfsel];
                vf.q[1] = *(const uint4*)&Vs[d + ks * 32 + halfsel + 16];
                o[j] = __builtin_amdgcn_wmma_f32_16x16x32_bf16(
                    false, pf[ks].v, false, vf.v, (short)0, o[j], false, false);
            }
        }
    }

    // Normalize and store O as bf16 [B,T,C]
    float inv[8];
#pragma unroll
    for (int r = 0; r < 8; ++r) inv[r] = 1.0f / lrow[r];
#pragma unroll
    for (int j = 0; j < 4; ++j)
#pragma unroll
        for (int r = 0; r < 8; ++r) {
            int t   = qRow0 + rowOff + r;
            int col = h * D + j * 16 + l15;
            Og[((size_t)b * T + t) * C + col] = f2bf(o[j][r] * inv[r]);
        }
}

// ---------------------------------------------------------------------------
// Host orchestration
// ---------------------------------------------------------------------------
extern "C" void kernel_launch(void* const* d_in, const int* in_sizes, int n_in,
                              void* d_out, int out_size, void* d_ws, size_t ws_size,
                              hipStream_t stream) {
    (void)in_sizes; (void)n_in; (void)out_size; (void)ws_size;

    const int Bb = 2, T = 2048, C = 1024, DL = 64;
    const int MT = Bb * T;                       // 4096 rows

    const float* x  = (const float*)d_in[0];
    const float* Wq = (const float*)d_in[1];
    const float* bq = (const float*)d_in[2];
    const float* Wk = (const float*)d_in[3];
    const float* bk = (const float*)d_in[4];
    const float* Wv = (const float*)d_in[5];
    const float* bv = (const float*)d_in[6];
    const float* Wd = (const float*)d_in[7];
    const float* bd = (const float*)d_in[8];
    const float* Wu = (const float*)d_in[9];
    const float* bu = (const float*)d_in[10];
    const float* Wo = (const float*)d_in[11];
    const float* bo = (const float*)d_in[12];

    // Workspace bump allocator (256B aligned)
    char*  ws  = (char*)d_ws;
    size_t off = 0;
    auto alloc = [&](size_t bytes) -> unsigned short* {
        unsigned short* p = (unsigned short*)(ws + off);
        off = (off + bytes + 255) & ~(size_t)255;
        return p;
    };

    unsigned short* xb  = alloc((size_t)MT * C * 2);   // x bf16; reused as attn out
    unsigned short* wqb = alloc((size_t)C * C * 2);
    unsigned short* wkb = alloc((size_t)C * C * 2);
    unsigned short* wvb = alloc((size_t)C * C * 2);
    unsigned short* wob = alloc((size_t)C * C * 2);
    unsigned short* wdb = alloc((size_t)C * DL * 2);
    unsigned short* wub = alloc((size_t)DL * C * 2);
    unsigned short* Qb  = alloc((size_t)MT * C * 2);
    unsigned short* Kcb = alloc((size_t)MT * C * 2);   // K proj; reused for K-up
    unsigned short* Vcb = alloc((size_t)MT * C * 2);   // V proj; reused for V-up
    unsigned short* Kdb = alloc((size_t)MT * DL * 2);
    unsigned short* Vdb = alloc((size_t)MT * DL * 2);

    // 1) fp32 -> bf16 conversions
    auto cvt = [&](const float* in, unsigned short* out, int n) {
        int blocks = (n + 255) / 256;
        if (blocks > 16384) blocks = 16384;
        cvt_f32_to_bf16<<<blocks, 256, 0, stream>>>(in, out, n);
    };
    cvt(x,  xb,  MT * C);
    cvt(Wq, wqb, C * C);
    cvt(Wk, wkb, C * C);
    cvt(Wv, wvb, C * C);
    cvt(Wo, wob, C * C);
    cvt(Wd, wdb, C * DL);
    cvt(Wu, wub, DL * C);

    // 2) Q/K/V projections  [MT,C] x [C,C]; 1/sqrt(D) folded into Q
    dim3 gBig(C / 128, MT / 128);                      // (8, 32)
    gemm_bf16_wmma<128, 128, 32, 64, true>
        <<<gBig, 256, 0, stream>>>(xb, wqb, bq, Qb, 0.125f, MT, C, C);
    gemm_bf16_wmma<128, 128, 32, 64, true>
        <<<gBig, 256, 0, stream>>>(xb, wkb, bk, Kcb, 1.0f, MT, C, C);
    gemm_bf16_wmma<128, 128, 32, 64, true>
        <<<gBig, 256, 0, stream>>>(xb, wvb, bv, Vcb, 1.0f, MT, C, C);

    // 3) Latent down-projection  [MT,C] x [C,64]
    dim3 gDown(DL / 64, MT / 64);                      // (1, 64)
    gemm_bf16_wmma<64, 64, 32, 32, true>
        <<<gDown, 128, 0, stream>>>(Kcb, wdb, bd, Kdb, 1.0f, MT, DL, C);
    gemm_bf16_wmma<64, 64, 32, 32, true>
        <<<gDown, 128, 0, stream>>>(Vcb, wdb, bd, Vdb, 1.0f, MT, DL, C);

    // 4) Latent up-projection  [MT,64] x [64,C]  (overwrites Kcb/Vcb)
    gemm_bf16_wmma<128, 128, 32, 64, true>
        <<<gBig, 256, 0, stream>>>(Kdb, wub, bu, Kcb, 1.0f, MT, C, DL);
    gemm_bf16_wmma<128, 128, 32, 64, true>
        <<<gBig, 256, 0, stream>>>(Vdb, wub, bu, Vcb, 1.0f, MT, C, DL);

    // 5) Attention: B*H*(T/64) = 1024 workgroups; output into xb
    mla_attention<<<Bb * 16 * (T / 64), 128, 0, stream>>>(Qb, Kcb, Vcb, xb);

    // 6) Output projection -> fp32 d_out
    gemm_bf16_wmma<128, 128, 32, 64, false>
        <<<gBig, 256, 0, stream>>>(xb, wob, bo, d_out, 1.0f, MT, C, C);
}
